// MambaLayer_68745246539920
// MI455X (gfx1250) — matrix operand order repaired and checked by reading, not compiled, not run on hardware
//
#include <hip/hip_runtime.h>
#include <hip/hip_bf16.h>

typedef __attribute__((ext_vector_type(16))) _Float16 v16h;
typedef __attribute__((ext_vector_type(8)))  _Float16 v8h;
typedef __attribute__((ext_vector_type(8)))  float    v8f;

#define DIMC 96
#define DINNER 192
#define DSTATE 16
#define DTRANK 6
#define DCONV 4
#define LSEQ 32768
#define NPROJ 384          // 2*DINNER
#define NXDBL 48           // DTRANK+2*DSTATE=38 padded to 48 (3 wmma n-tiles)
#define KMAX 192           // largest GEMM K (x_proj / out_proj)

// ---------------------------------------------------------------------------
// Weight prep: f32 -> f16, with x_proj padded from 38 to 48 rows (zeros).
// ---------------------------------------------------------------------------
__global__ void prep_weights_kernel(const float* __restrict__ in_proj_w,
                                    const float* __restrict__ x_proj_w,
                                    const float* __restrict__ out_proj_w,
                                    _Float16* __restrict__ win,   // (384,96)
                                    _Float16* __restrict__ wxp,   // (48,192)
                                    _Float16* __restrict__ wout)  // (96,192)
{
    int tid = blockIdx.x * blockDim.x + threadIdx.x;
    const int n_in  = NPROJ * DIMC;          // 36864
    const int n_xp  = NXDBL * DINNER;        // 9216
    const int n_out = DIMC * DINNER;         // 18432
    if (tid < n_in) {
        win[tid] = (_Float16)in_proj_w[tid];
    } else if (tid < n_in + n_xp) {
        int i = tid - n_in;
        int n = i / DINNER, k = i % DINNER;
        wxp[i] = (n < (DTRANK + 2 * DSTATE)) ? (_Float16)x_proj_w[n * DINNER + k]
                                             : (_Float16)0.0f;
    } else if (tid < n_in + n_xp + n_out) {
        int i = tid - n_in - n_xp;
        wout[i] = (_Float16)out_proj_w[i];
    }
}

// ---------------------------------------------------------------------------
// LayerNorm over C=96 per token + NCDHW->(L,C) transpose, output f16 for WMMA.
// Token l = h*1024 + w*32 + d ; source index = c*32768 + d*1024 + h*32 + w.
// Thread index chosen so source reads are coalesced (w fastest).
// ---------------------------------------------------------------------------
__global__ void layernorm_kernel(const float* __restrict__ x,
                                 const float* __restrict__ nw,
                                 const float* __restrict__ nb,
                                 _Float16* __restrict__ xnorm) // (L,96)
{
    int tid = blockIdx.x * blockDim.x + threadIdx.x;
    if (tid >= LSEQ) return;
    int w = tid & 31, h = (tid >> 5) & 31, d = tid >> 10;
    int l = h * 1024 + w * 32 + d;
    size_t src = (size_t)d * 1024 + h * 32 + w;
    float s = 0.f, s2 = 0.f;
    for (int c = 0; c < DIMC; ++c) {
        float v = x[(size_t)c * (size_t)LSEQ + src];
        s += v; s2 += v * v;
    }
    float mu = s * (1.0f / DIMC);
    float var = s2 * (1.0f / DIMC) - mu * mu;
    float rs = rsqrtf(var + 1e-5f);
    _Float16* dst = xnorm + (size_t)l * DIMC;
    for (int c = 0; c < DIMC; ++c) {
        float v = x[(size_t)c * (size_t)LSEQ + src];
        dst[c] = (_Float16)(((v - mu) * rs) * nw[c] + nb[c]);
    }
}

// ---------------------------------------------------------------------------
// WMMA GEMM with async-LDS weight staging:
//   C(M,N) f32 = A(M,K) f16  x  W(N,K)^T f16
// Block = 128 threads (4 waves). Each block owns one 16-col n-tile and 4
// m-tiles. The shared 16xK B tile is staged into LDS once per block with
// GLOBAL_LOAD_ASYNC_TO_LDS_B128 (ASYNCcnt), then all waves read B fragments
// from LDS while streaming A fragments from global.
// Fragment layouts per CDNA5 ISA 7.12.2 (wave32).
// ---------------------------------------------------------------------------
__global__ void wmma_gemm_kernel(const _Float16* __restrict__ A,
                                 const _Float16* __restrict__ W,
                                 float* __restrict__ C,
                                 int M, int N, int K)
{
    __shared__ __align__(16) _Float16 sB[16 * KMAX];

    const int nTile = blockIdx.x;
    const int nBase = nTile * 16;

    // --- async stage: 16 rows x K f16 of W into LDS, 16B chunks ------------
    {
        const int chunksPerRow = K >> 3;              // (K*2)/16
        const int totalChunks  = 16 * chunksPerRow;
        const unsigned int sbase = (unsigned int)(uintptr_t)(&sB[0]);
        for (int c = threadIdx.x; c < totalChunks; c += blockDim.x) {
            int row = c / chunksPerRow;
            int e   = (c - row * chunksPerRow) << 3;  // f16 element offset
            const _Float16* g = W + (size_t)(nBase + row) * K + e;
            unsigned int lds = sbase + (unsigned int)(row * K + e) * 2u;
            asm volatile("global_load_async_to_lds_b128 %0, %1, off"
                         :: "v"(lds),
                            "v"((unsigned long long)(uintptr_t)g)
                         : "memory");
        }
        asm volatile("s_wait_asynccnt 0" ::: "memory");
    }
    __syncthreads();

    const int lane = threadIdx.x & 31;
    const int wave = threadIdx.x >> 5;           // 4 waves / block
    const int mTile = blockIdx.y * 4 + wave;
    const int mBase = mTile * 16;
    if (mBase >= M) return;

    const bool hi = lane >= 16;
    const int  l15 = lane & 15;
    const _Float16* arow = A + (size_t)(mBase + l15) * K;
    const _Float16* brow = sB + l15 * K;         // B column l15, from LDS

    v8f acc = {};
    for (int k0 = 0; k0 < K; k0 += 32) {
        union { v16h v; v8h h[2]; } a, b;
        int aoff = k0 + (hi ? 8 : 0);
        a.h[0] = *(const v8h*)(arow + aoff);        // K: aoff .. aoff+7
        a.h[1] = *(const v8h*)(arow + aoff + 16);   // K: aoff+16 .. aoff+23
        int boff = k0 + (hi ? 8 : 0);
        b.h[0] = *(const v8h*)(brow + boff);        // K: boff .. boff+7  (ds)
        b.h[1] = *(const v8h*)(brow + boff + 16);   // K: boff+16 .. boff+23
        acc = __builtin_amdgcn_wmma_f32_16x16x32_f16(
            false, a.v, false, b.v, (short)0, acc, false, false);
    }

    float* crow = C + (size_t)(mBase + (hi ? 8 : 0)) * N + nBase + l15;
#pragma unroll
    for (int v = 0; v < 8; ++v) crow[(size_t)v * N] = acc[v];
}

// ---------------------------------------------------------------------------
// Depthwise causal conv (k=4) over L, + bias, + SiLU. xs = xz[:, 0:192].
// ---------------------------------------------------------------------------
__global__ void conv_silu_kernel(const float* __restrict__ xz,     // (L,384)
                                 const float* __restrict__ conv_w, // (192,4)
                                 const float* __restrict__ conv_b,
                                 float* __restrict__ xc,           // (L,192) f32
                                 _Float16* __restrict__ xc16)      // (L,192) f16
{
    int tid = blockIdx.x * blockDim.x + threadIdx.x;
    if (tid >= LSEQ * DINNER) return;
    int d = tid % DINNER;
    int l = tid / DINNER;
    float acc = conv_b[d];
#pragma unroll
    for (int j = 0; j < DCONV; ++j) {
        int ls = l - (DCONV - 1) + j;
        if (ls >= 0) acc += conv_w[d * DCONV + j] * xz[(size_t)ls * NPROJ + d];
    }
    float sv = acc / (1.0f + __expf(-acc));   // silu
    xc[(size_t)l * DINNER + d]   = sv;
    xc16[(size_t)l * DINNER + d] = (_Float16)sv;
}

// ---------------------------------------------------------------------------
// delta = softplus(dt @ dt_proj_w^T + b) ; dt = x_dbl[:, 0:6].
// ---------------------------------------------------------------------------
__global__ void delta_kernel(const float* __restrict__ xdbl, // (L,48)
                             const float* __restrict__ dtw,  // (192,6)
                             const float* __restrict__ dtb,
                             float* __restrict__ delta)      // (L,192)
{
    int tid = blockIdx.x * blockDim.x + threadIdx.x;
    if (tid >= LSEQ * DINNER) return;
    int d = tid % DINNER;
    int l = tid / DINNER;
    const float* dt = xdbl + (size_t)l * NXDBL;
    float acc = dtb[d];
#pragma unroll
    for (int r = 0; r < DTRANK; ++r) acc += dt[r] * dtw[d * DTRANK + r];
    float sp = (acc > 20.0f) ? acc : log1pf(__expf(acc));
    delta[(size_t)l * DINNER + d] = sp;
}

// ---------------------------------------------------------------------------
// Selective scan. Channels d are independent: 12 blocks x (16 d x 16 s).
// h[d,s] lives in one register; y reduced over s with intra-wave shfl_xor.
// Fused: + xc*D, * silu(z), cast f16 for out_proj WMMA.
// ---------------------------------------------------------------------------
__global__ void scan_kernel(const float* __restrict__ delta, // (L,192)
                            const float* __restrict__ xdbl,  // (L,48): B at 6, C at 22
                            const float* __restrict__ xc,    // (L,192)
                            const float* __restrict__ xz,    // (L,384): z at +192
                            const float* __restrict__ A_log, // (192,16)
                            const float* __restrict__ Dp,    // (192,)
                            _Float16* __restrict__ yfin)     // (L,192)
{
    int t = threadIdx.x;            // 256
    int s = t & 15;
    int dl = t >> 4;                // 0..15
    int d = blockIdx.x * 16 + dl;   // 0..191
    float a = -__expf(A_log[d * DSTATE + s]);
    float dpar = Dp[d];
    float h = 0.0f;
    for (int l = 0; l < LSEQ; ++l) {
        if (((l & 63) == 0) && (l + 64 < LSEQ))
            __builtin_prefetch(delta + (size_t)(l + 64) * DINNER + d, 0, 1);
        float dlt = delta[(size_t)l * DINNER + d];
        float xcv = xc[(size_t)l * DINNER + d];
        float bm  = xdbl[(size_t)l * NXDBL + DTRANK + s];
        float cm  = xdbl[(size_t)l * NXDBL + DTRANK + DSTATE + s];
        float dA = __expf(dlt * a);
        h = dA * h + dlt * bm * xcv;
        float contrib = h * cm;
        contrib += __shfl_xor(contrib, 1);
        contrib += __shfl_xor(contrib, 2);
        contrib += __shfl_xor(contrib, 4);
        contrib += __shfl_xor(contrib, 8);
        if (s == 0) {
            float y  = contrib + xcv * dpar;
            float zv = xz[(size_t)l * NPROJ + DINNER + d];
            float sz = zv / (1.0f + __expf(-zv));
            yfin[(size_t)l * DINNER + d] = (_Float16)(y * sz);
        }
    }
}

// ---------------------------------------------------------------------------
// Scatter (L,96) back to (C,D,H,W):  out[c,d,h,w] = y96[l=h*1024+w*32+d][c]
// ---------------------------------------------------------------------------
__global__ void scatter_kernel(const float* __restrict__ y96,
                               float* __restrict__ out)
{
    int tid = blockIdx.x * blockDim.x + threadIdx.x;
    if (tid >= LSEQ * DIMC) return;
    int c = tid % DIMC;
    int l = tid / DIMC;
    int d = l & 31, w = (l >> 5) & 31, h = l >> 10;
    out[(size_t)c * LSEQ + (size_t)d * 1024 + h * 32 + w] = y96[(size_t)l * DIMC + c];
}

// ---------------------------------------------------------------------------
static inline char* ws_take(char* base, size_t& off, size_t bytes) {
    char* p = base + off;
    off = (off + bytes + 255) & ~(size_t)255;
    return p;
}

extern "C" void kernel_launch(void* const* d_in, const int* in_sizes, int n_in,
                              void* d_out, int out_size, void* d_ws, size_t ws_size,
                              hipStream_t stream) {
    const float* x        = (const float*)d_in[0];
    const float* norm_w   = (const float*)d_in[1];
    const float* norm_b   = (const float*)d_in[2];
    const float* in_proj  = (const float*)d_in[3];
    const float* conv_w   = (const float*)d_in[4];
    const float* conv_b   = (const float*)d_in[5];
    const float* x_proj   = (const float*)d_in[6];
    const float* dt_w     = (const float*)d_in[7];
    const float* dt_b     = (const float*)d_in[8];
    const float* A_log    = (const float*)d_in[9];
    const float* D_param  = (const float*)d_in[10];
    const float* out_proj = (const float*)d_in[11];
    float* out = (float*)d_out;

    char* ws = (char*)d_ws;
    size_t off = 0;
    _Float16* win   = (_Float16*)ws_take(ws, off, (size_t)NPROJ * DIMC * 2);
    _Float16* wxp   = (_Float16*)ws_take(ws, off, (size_t)NXDBL * DINNER * 2);
    _Float16* wout  = (_Float16*)ws_take(ws, off, (size_t)DIMC * DINNER * 2);
    _Float16* xnorm = (_Float16*)ws_take(ws, off, (size_t)LSEQ * DIMC * 2);
    float*    xz    = (float*)   ws_take(ws, off, (size_t)LSEQ * NPROJ * 4);
    float*    xc    = (float*)   ws_take(ws, off, (size_t)LSEQ * DINNER * 4);
    _Float16* xc16  = (_Float16*)ws_take(ws, off, (size_t)LSEQ * DINNER * 2);
    float*    xdbl  = (float*)   ws_take(ws, off, (size_t)LSEQ * NXDBL * 4);
    float*    delta = (float*)   ws_take(ws, off, (size_t)LSEQ * DINNER * 4);
    _Float16* yfin  = (_Float16*)ws_take(ws, off, (size_t)LSEQ * DINNER * 2);
    float*    y96   = (float*)   ws_take(ws, off, (size_t)LSEQ * DIMC * 4);

    // 1. weights -> f16 (x_proj zero-padded to 48 rows)
    {
        int total = NPROJ * DIMC + NXDBL * DINNER + DIMC * DINNER;
        prep_weights_kernel<<<(total + 255) / 256, 256, 0, stream>>>(
            in_proj, x_proj, out_proj, win, wxp, wout);
    }
    // 2. layernorm + transpose -> xnorm f16 (L,96)
    layernorm_kernel<<<LSEQ / 256, 256, 0, stream>>>(x, norm_w, norm_b, xnorm);
    // 3. in_proj: (L,96) x (96,384) -> xz (L,384)
    wmma_gemm_kernel<<<dim3(NPROJ / 16, LSEQ / 64), 128, 0, stream>>>(
        xnorm, win, xz, LSEQ, NPROJ, DIMC);
    // 4. causal depthwise conv + SiLU -> xc f32/f16
    conv_silu_kernel<<<(LSEQ * DINNER + 255) / 256, 256, 0, stream>>>(
        xz, conv_w, conv_b, xc, xc16);
    // 5. x_proj: (L,192) x (192,48) -> x_dbl (L,48)
    wmma_gemm_kernel<<<dim3(NXDBL / 16, LSEQ / 64), 128, 0, stream>>>(
        xc16, wxp, xdbl, LSEQ, NXDBL, DINNER);
    // 6. dt_proj (K=6) + softplus -> delta (L,192)
    delta_kernel<<<(LSEQ * DINNER + 255) / 256, 256, 0, stream>>>(
        xdbl, dt_w, dt_b, delta);
    // 7. selective scan (12 independent channel blocks) -> yfin f16 (L,192)
    scan_kernel<<<DINNER / 16, 256, 0, stream>>>(
        delta, xdbl, xc, xz, A_log, D_param, yfin);
    // 8. out_proj: (L,192) x (192,96) -> y96 (L,96)
    wmma_gemm_kernel<<<dim3(DIMC / 16, LSEQ / 64), 128, 0, stream>>>(
        yfin, wout, y96, LSEQ, DIMC, DINNER);
    // 9. scatter back to (C,D,H,W)
    scatter_kernel<<<(LSEQ * DIMC + 255) / 256, 256, 0, stream>>>(y96, out);
}
